// HistogramMatching_28226525070046
// MI455X (gfx1250) — compile-verified, matrix operand order checked by reading, not run on hardware
//
#include <hip/hip_runtime.h>

// ---------------------------------------------------------------------------
// Histogram matching on MI455X (gfx1250, wave32).
//
// Roofline: ~9.4 MB total HBM traffic -> ~0.4 us floor at 23.3 TB/s.
// sigmoid(75*t) saturates for |t| > ~0.3, so each pixel's soft-histogram
// contribution is non-zero for only 3 bins -> 2 v_exp_f32 per pixel instead
// of 512 dense sigmoids. Per-lane LDS sub-histograms ([bin][lane] layout)
// make ds_add_f32 bank-conflict-free in wave32. The 32-partial reduction is
// done on the matrix core: D = ones(16x32) x B(32x16) via
// v_wmma_f32_16x16x32_bf16 (f32 accumulate). Streaming loads/stores use
// float4 (global_load_b128 / global_store_b128): 512 B per wave32 request.
// No TDM/async-to-LDS: data is consumed once, register-direct is optimal.
// ---------------------------------------------------------------------------

typedef __attribute__((ext_vector_type(16))) __bf16 v16bf;
typedef __attribute__((ext_vector_type(8)))  float  v8f;

#define HM_HW        65536          // H*W = 256*256
#define HM_ROWS      12             // B*C
#define HM_NHIST     24             // dst rows + ref rows
#define HM_CHUNKS    16             // blocks per row-image
#define HM_PIXBLK    (HM_HW / HM_CHUNKS)   // 4096 pixels per block

__global__ void hm_zero_kernel(float* __restrict__ hist) {
    hist[blockIdx.x * 256 + threadIdx.x] = 0.0f;
}

// Soft-histogram contribution of one pixel: bins k0-1, k0, k0+1 only.
__device__ __forceinline__ void hm_accum(float x, float* s_hist, int lane) {
    float v  = x * 255.0f;                      // v in [0, 255)
    float kf = floorf(v);
    int   k0 = (int)kf;
    k0 = k0 < 0 ? 0 : (k0 > 255 ? 255 : k0);
    float f  = v - kf;
    //   h[k0-1] = sig(75(f+1)) - sig(75 f)    ~= 1 - s1
    //   h[k0]   = sig(75 f)    - sig(75(f-1))  = s1 - s0
    //   h[k0+1] = sig(75(f-1)) - sig(75(f-2)) ~= s0
    float s1 = 1.0f / (1.0f + __expf(-75.0f * f));
    float s0 = 1.0f / (1.0f + __expf(-75.0f * (f - 1.0f)));
    if (k0 >= 1)   atomicAdd(&s_hist[(k0 - 1) * 32 + lane], 1.0f - s1);
    atomicAdd(&s_hist[k0 * 32 + lane], s1 - s0);
    if (k0 <= 254) atomicAdd(&s_hist[(k0 + 1) * 32 + lane], s0);
}

__global__ void hm_hist_kernel(const float* __restrict__ dst,
                               const float* __restrict__ ref,
                               float* __restrict__ hist) {
    // [bin][sub] layout, sub == lane: addresses (bin*32+lane)*4 never collide
    // within a wave32 (collision would need lane distance 32).
    __shared__ float s_hist[256 * 32];    // 32 KB of the 320 KB WGP pool

    const int tid    = threadIdx.x;
    const int lane   = tid & 31;
    const int wave   = tid >> 5;          // 8 waves / block
    const int rowimg = blockIdx.x;        // 0..11 dst rows, 12..23 ref rows
    const int chunk  = blockIdx.y;        // 0..15

    for (int i = tid; i < 256 * 32; i += 256) s_hist[i] = 0.0f;
    __syncthreads();

    const float* src = (rowimg < HM_ROWS ? dst + rowimg * HM_HW
                                         : ref + (rowimg - HM_ROWS) * HM_HW)
                       + chunk * HM_PIXBLK;
    const float4* src4 = reinterpret_cast<const float4*>(src);

    // 4096 px / block: 4 iterations of global_load_b128 per thread.
    for (int p = tid; p < HM_PIXBLK / 4; p += 256) {
        __builtin_prefetch(src4 + p + 256, 0, 1);   // global_prefetch_b8
        float4 px = src4[p];
        hm_accum(px.x, s_hist, lane);
        hm_accum(px.y, s_hist, lane);
        hm_accum(px.z, s_hist, lane);
        hm_accum(px.w, s_hist, lane);
    }
    __syncthreads();

    // Matrix-core reduction of the 32 sub-histograms:
    // A = ones(16x32) bf16, B(32x16) = sub-hist values for one 16-bin group,
    // D[m,n] = sum_k B[k,n]  (row 0 read out by lanes 0..15).
    v16bf a;
#pragma unroll
    for (int i = 0; i < 16; ++i) a[i] = (__bf16)1.0f;

    const int n     = lane & 15;   // B column (bin within group)
    const int khalf = lane >> 4;   // which half of K this lane supplies

    for (int g = wave; g < 16; g += 8) {          // 16 bin-groups over 8 waves
        v16bf b;
        const float* col = &s_hist[(g * 16 + n) * 32 + khalf * 16];
#pragma unroll
        for (int j = 0; j < 16; ++j) b[j] = (__bf16)col[j];
        v8f c = {};
        c = __builtin_amdgcn_wmma_f32_16x16x32_bf16(
                /*neg_a=*/false, a, /*neg_b=*/false, b,
                /*c_mod=*/(short)0, c, /*reuse_a=*/false, /*reuse_b=*/false);
        if (lane < 16)   // D VGPR0, lanes 0..15 hold M=0, N=lane
            atomicAdd(&hist[rowimg * 256 + g * 16 + lane], c[0]);
    }
}

// Per row: total = sum|h|, normalize, inclusive cumsum -> CDF.
__global__ void hm_cdf_kernel(const float* __restrict__ hist,
                              float* __restrict__ cdf) {
    __shared__ float s[256];
    __shared__ float red[256];
    const int b = blockIdx.x, j = threadIdx.x;
    float h = hist[b * 256 + j];
    red[j] = fabsf(h);
    __syncthreads();
    for (int off = 128; off > 0; off >>= 1) {
        if (j < off) red[j] += red[j + off];
        __syncthreads();
    }
    float total = red[0] < 1e-12f ? 1e-12f : red[0];
    s[j] = h / total;
    __syncthreads();
    // Hillis-Steele inclusive scan over 256 bins
    for (int off = 1; off < 256; off <<= 1) {
        float add = (j >= off) ? s[j - off] : 0.0f;
        __syncthreads();
        s[j] += add;
        __syncthreads();
    }
    cdf[b * 256 + j] = s[j];
}

// table[r,j] = clip( sum_i (cdf_dst[r,j] >= cdf_ref[r,i]) - 1, 0, 255 )
__global__ void hm_table_kernel(const float* __restrict__ cdf,
                                float* __restrict__ table) {
    __shared__ float sref[256];
    const int b = blockIdx.x, j = threadIdx.x;
    sref[j] = cdf[(HM_ROWS + b) * 256 + j];    // ref CDFs live in rows 12..23
    __syncthreads();
    float dj  = cdf[b * 256 + j];
    int   cnt = 0;
#pragma unroll 8
    for (int i = 0; i < 256; ++i)
        cnt += (dj - sref[i] >= 0.0f) ? 1 : 0;
    float t = (float)(cnt - 1);
    t = fminf(fmaxf(t, 0.0f), 255.0f);
    table[b * 256 + j] = t;
}

// out = table[b*c][int(255*dst)] / 255   (reference uses row index b*c).
// Vectorized: one float4 load + one float4 store per thread (b128 each).
__global__ void hm_apply_kernel(const float* __restrict__ dst,
                                const float* __restrict__ table,
                                float* __restrict__ out, int total4) {
    int i = blockIdx.x * blockDim.x + threadIdx.x;
    if (i >= total4) return;
    int pix = i * 4;                 // 4 consecutive pixels, same (b,c) row
    int bc  = pix / HM_HW;           // 0..11
    int b   = bc / 3, c = bc % 3;
    const float* trow = table + (b * c) * 256;   // reference indexing quirk

    float4 px = reinterpret_cast<const float4*>(dst)[i];
    float4 r;
    {
        int k = (int)(px.x * 255.0f); k = k < 0 ? 0 : (k > 255 ? 255 : k);
        r.x = trow[k] * (1.0f / 255.0f);
    }
    {
        int k = (int)(px.y * 255.0f); k = k < 0 ? 0 : (k > 255 ? 255 : k);
        r.y = trow[k] * (1.0f / 255.0f);
    }
    {
        int k = (int)(px.z * 255.0f); k = k < 0 ? 0 : (k > 255 ? 255 : k);
        r.z = trow[k] * (1.0f / 255.0f);
    }
    {
        int k = (int)(px.w * 255.0f); k = k < 0 ? 0 : (k > 255 ? 255 : k);
        r.w = trow[k] * (1.0f / 255.0f);
    }
    reinterpret_cast<float4*>(out)[i] = r;
}

extern "C" void kernel_launch(void* const* d_in, const int* in_sizes, int n_in,
                              void* d_out, int out_size, void* d_ws, size_t ws_size,
                              hipStream_t stream) {
    const float* dst = (const float*)d_in[0];
    const float* ref = (const float*)d_in[1];
    float* out   = (float*)d_out;
    float* hist  = (float*)d_ws;              // 24 x 256 f32
    float* cdf   = hist + HM_NHIST * 256;     // 24 x 256 f32
    float* table = cdf  + HM_NHIST * 256;     // 12 x 256 f32

    hm_zero_kernel<<<HM_NHIST, 256, 0, stream>>>(hist);
    hm_hist_kernel<<<dim3(HM_NHIST, HM_CHUNKS), 256, 0, stream>>>(dst, ref, hist);
    hm_cdf_kernel<<<HM_NHIST, 256, 0, stream>>>(hist, cdf);
    hm_table_kernel<<<HM_ROWS, 256, 0, stream>>>(cdf, table);

    const int total4 = (HM_ROWS * HM_HW) / 4;   // 196608 float4s
    hm_apply_kernel<<<(total4 + 255) / 256, 256, 0, stream>>>(dst, table, out, total4);
}
